// TransmutationModel_60541859004419
// MI455X (gfx1250) — matrix operand order repaired and checked by reading, not compiled, not run on hardware
//
#include <hip/hip_runtime.h>
#include <hip/hip_bf16.h>
#include <math.h>

typedef __bf16 bf16_t;
typedef __attribute__((ext_vector_type(16))) __bf16 v16bf;
typedef __attribute__((ext_vector_type(8)))  __bf16 v8bf;
typedef __attribute__((ext_vector_type(8)))  float  v8f;

#define D_MODEL_C 512
#define D_STATE_C 16
#define D_CONV_C  4
#define D_INNER_C 1024
#define DT_RANK_C 32
#define N_HEADS_C 8
#define HD_C      64
#define VOCAB_C   32000
#define NB_C      2
#define SRC_C     1024
#define TGT_C     512
#define MENC_C    (NB_C * SRC_C)   // 2048
#define MDEC_C    (NB_C * TGT_C)   // 1024
#define EMB_SCALE 22.62741699796952f  // sqrt(512)

__device__ __forceinline__ v16bf load_frag(const bf16_t* p, int off)
{
    v8bf lo = *(const v8bf*)p;
    v8bf hi = *(const v8bf*)(p + off);
    v16bf r;
    #pragma unroll
    for (int e = 0; e < 8; ++e) { r[e] = lo[e]; r[e + 8] = hi[e]; }
    return r;
}

__device__ __forceinline__ float apply_act(float val, int act)
{
    if (act == 2)        // softplus (stable)
        val = (val > 0.f) ? (val + log1pf(expf(-val))) : log1pf(expf(val));
    else if (act == 3)   // exact gelu
        val = 0.5f * val * (1.f + erff(val * 0.70710678118654752f));
    return val;
}

// ---------------------------------------------------------------------------
// WMMA GEMM: C = act( A[M,K](bf16) x B[N,K](bf16 row-major) + bias )
// f32 accumulate; optional fused bf16 copy of C; optional residual accumulate
// (C[i] += val).  Optional (b,h) batching via gridDim.y: batch z -> b=z>>3,
// h=z&7, operand offsets b*s?b + h*s?h (C and Cbf share sC offsets).
// One 32x32 tile per wave: 2 A-frags x 2 B-frags -> 4 WMMAs per 32-k step.
// Fragment layouts per CDNA5 ISA 16-bit striping (A: k=half*8+e / +16;
// B: k=half*8+e / +16 (same as A); C: row=half*8+v, col=lane&15).
// ---------------------------------------------------------------------------
__global__ __launch_bounds__(256)
void k_gemm(const bf16_t* __restrict__ A, const bf16_t* __restrict__ B,
            const float* __restrict__ bias, float* __restrict__ C,
            bf16_t* __restrict__ Cbf,
            int M, int N, int K, int lda, int ldb, int ldc, int act, int accum,
            long sAb, long sAh, long sBb, long sBh, long sCb, long sCh)
{
    const int lane = threadIdx.x & 31;
    const int wave = threadIdx.x >> 5;
    const int ntn  = N >> 5;
    const int tile = blockIdx.x * 8 + wave;
    if (tile >= (M >> 5) * ntn) return;
    const int z = blockIdx.y;
    A += (long)(z >> 3) * sAb + (long)(z & 7) * sAh;
    B += (long)(z >> 3) * sBb + (long)(z & 7) * sBh;
    const long offC = (long)(z >> 3) * sCb + (long)(z & 7) * sCh;
    C += offC;
    if (Cbf) Cbf += offC;

    const int mt   = tile / ntn;
    const int nt   = tile % ntn;
    const int half = lane >> 4;
    const int l15  = lane & 15;
    const int n0   = (nt << 5) + l15;

    const bf16_t* ap0 = A + (size_t)((mt << 5) + l15) * lda + half * 8;
    const bf16_t* ap1 = ap0 + (size_t)16 * lda;
    const bf16_t* bp0 = B + (size_t)n0 * ldb + half * 8;
    const bf16_t* bp1 = bp0 + (size_t)16 * ldb;

    v8f acc00 = {}, acc01 = {}, acc10 = {}, acc11 = {};
    #pragma unroll 2
    for (int kb = 0; kb < K; kb += 32) {
        const v16bf af0 = load_frag(ap0, 16);
        const v16bf af1 = load_frag(ap1, 16);
        const v16bf bf0 = load_frag(bp0, 16);
        const v16bf bf1 = load_frag(bp1, 16);
        acc00 = __builtin_amdgcn_wmma_f32_16x16x32_bf16(false, af0, false, bf0, (short)0, acc00, false, false);
        acc01 = __builtin_amdgcn_wmma_f32_16x16x32_bf16(false, af0, false, bf1, (short)0, acc01, false, false);
        acc10 = __builtin_amdgcn_wmma_f32_16x16x32_bf16(false, af1, false, bf0, (short)0, acc10, false, false);
        acc11 = __builtin_amdgcn_wmma_f32_16x16x32_bf16(false, af1, false, bf1, (short)0, acc11, false, false);
        ap0 += 32; ap1 += 32; bp0 += 32; bp1 += 32;
    }

    const float bv0 = bias ? bias[n0]      : 0.f;
    const float bv1 = bias ? bias[n0 + 16] : 0.f;
    #pragma unroll
    for (int v = 0; v < 8; ++v) {
        const int r0 = (mt << 5) + half * 8 + v;
        const int r1 = r0 + 16;
        float v00 = apply_act(acc00[v] + bv0, act);
        float v01 = apply_act(acc01[v] + bv1, act);
        float v10 = apply_act(acc10[v] + bv0, act);
        float v11 = apply_act(acc11[v] + bv1, act);
        const size_t i00 = (size_t)r0 * ldc + n0;
        const size_t i10 = (size_t)r1 * ldc + n0;
        if (accum) {
            C[i00] += v00; C[i00 + 16] += v01;
            C[i10] += v10; C[i10 + 16] += v11;
        } else {
            C[i00] = v00; C[i00 + 16] = v01;
            C[i10] = v10; C[i10 + 16] = v11;
            if (Cbf) {
                Cbf[i00] = (bf16_t)v00; Cbf[i00 + 16] = (bf16_t)v01;
                Cbf[i10] = (bf16_t)v10; Cbf[i10 + 16] = (bf16_t)v11;
            }
        }
    }
}

// ---------------------------------------------------------------------------
__global__ __launch_bounds__(256)
void k_cast(const float* __restrict__ in, bf16_t* __restrict__ out, long n)
{
    long i = (long)blockIdx.x * 256 + threadIdx.x;
    const long stride = (long)gridDim.x * 256;
    for (; i < n; i += stride) out[i] = (bf16_t)in[i];
}

__global__ __launch_bounds__(256)
void k_embed(const int* __restrict__ ids, const float* __restrict__ emb,
             float* __restrict__ out)
{
    const int t = blockIdx.x;
    const int c = threadIdx.x;
    const float* e = emb + (size_t)ids[t] * D_MODEL_C;
    float* o = out + (size_t)t * D_MODEL_C;
    o[c]       = e[c]       * EMB_SCALE;
    o[c + 256] = e[c + 256] * EMB_SCALE;
}

// LayerNorm over 512 cols; emits bf16 (consumed only as GEMM A operands)
__global__ __launch_bounds__(256)
void k_layernorm(const float* __restrict__ x, const float* __restrict__ g,
                 const float* __restrict__ b, bf16_t* __restrict__ out)
{
    __shared__ float red[256];
    const int row = blockIdx.x, t = threadIdx.x;
    const float* xr = x + (size_t)row * D_MODEL_C;
    float v0 = xr[t], v1 = xr[t + 256];
    red[t] = v0 + v1;
    __syncthreads();
    for (int s = 128; s > 0; s >>= 1) { if (t < s) red[t] += red[t + s]; __syncthreads(); }
    const float mean = red[0] * (1.f / D_MODEL_C);
    __syncthreads();
    const float d0 = v0 - mean, d1 = v1 - mean;
    red[t] = d0 * d0 + d1 * d1;
    __syncthreads();
    for (int s = 128; s > 0; s >>= 1) { if (t < s) red[t] += red[t + s]; __syncthreads(); }
    const float rstd = rsqrtf(red[0] * (1.f / D_MODEL_C) + 1e-5f);
    bf16_t* o = out + (size_t)row * D_MODEL_C;
    o[t]       = (bf16_t)(d0 * rstd * g[t]       + b[t]);
    o[t + 256] = (bf16_t)(d1 * rstd * g[t + 256] + b[t + 256]);
}

// causal depthwise conv over xc = xz[:, :1024] (row stride 2048) + SiLU
__global__ __launch_bounds__(256)
void k_conv_silu(const float* __restrict__ xz, const float* __restrict__ cw,
                 const float* __restrict__ cb, float* __restrict__ out,
                 bf16_t* __restrict__ outb, int L, long total)
{
    const long i = (long)blockIdx.x * 256 + threadIdx.x;
    if (i >= total) return;
    const int d = (int)(i & (D_INNER_C - 1));
    const long t = i >> 10;
    const int l = (int)(t % L);
    float acc = cb[d];
    #pragma unroll
    for (int k = 0; k < D_CONV_C; ++k) {
        const int ls = l - (D_CONV_C - 1) + k;
        if (ls >= 0)
            acc += xz[(t + (ls - l)) * (2 * D_INNER_C) + d] * cw[d * D_CONV_C + k];
    }
    acc = acc / (1.f + expf(-acc));
    out[i] = acc;
    outb[i] = (bf16_t)acc;
}

// selective scan: one thread per (b, d), 16 states in registers, sequential L
__global__ __launch_bounds__(256)
void k_scan(const float* __restrict__ dt, const float* __restrict__ dbc,
            const float* __restrict__ xc, const float* __restrict__ A_log,
            const float* __restrict__ Dp, float* __restrict__ y, int L)
{
    const int d = blockIdx.x * 256 + threadIdx.x;
    const int b = blockIdx.y;
    float A[D_STATE_C], h[D_STATE_C];
    #pragma unroll
    for (int n = 0; n < D_STATE_C; ++n) {
        A[n] = -expf(A_log[d * D_STATE_C + n]);
        h[n] = 0.f;
    }
    const float Dd = Dp[d];
    for (int l = 0; l < L; ++l) {
        const size_t t = (size_t)b * L + l;
        const float dv = dt[t * D_INNER_C + d];
        const float xv = xc[t * D_INNER_C + d];
        const float* bc = dbc + t * 64;     // [dt(32) | B(16) | C(16)]
        const float dx = dv * xv;
        float acc = 0.f;
        #pragma unroll
        for (int n = 0; n < D_STATE_C; ++n) {
            h[n] = expf(dv * A[n]) * h[n] + dx * bc[DT_RANK_C + n];
            acc += h[n] * bc[DT_RANK_C + D_STATE_C + n];
        }
        y[t * D_INNER_C + d] = acc + Dd * xv;
    }
}

// out_bf = y * silu(z), z = xz[:, 1024:]
__global__ __launch_bounds__(256)
void k_gate(const float* __restrict__ y, const float* __restrict__ xz,
            bf16_t* __restrict__ out, long total)
{
    const long i = (long)blockIdx.x * 256 + threadIdx.x;
    if (i >= total) return;
    const long t = i >> 10;
    const int d = (int)(i & (D_INNER_C - 1));
    const float z = xz[t * (2 * D_INNER_C) + D_INNER_C + d];
    out[i] = (bf16_t)(y[i] * (z / (1.f + expf(-z))));
}

// softmax over rows of 1024 (scale 1/sqrt(64) folded in); f32 + bf16 outputs
__global__ __launch_bounds__(256)
void k_softmax(float* __restrict__ s, bf16_t* __restrict__ sb)
{
    __shared__ float red[256];
    const size_t base = (size_t)blockIdx.x * SRC_C;
    float* r = s + base;
    const int t = threadIdx.x;
    float v[4], mx = -1e30f;
    #pragma unroll
    for (int j = 0; j < 4; ++j) {
        v[j] = r[t + j * 256] * 0.125f;
        mx = fmaxf(mx, v[j]);
    }
    red[t] = mx; __syncthreads();
    for (int st = 128; st > 0; st >>= 1) { if (t < st) red[t] = fmaxf(red[t], red[t + st]); __syncthreads(); }
    mx = red[0]; __syncthreads();
    float sum = 0.f;
    #pragma unroll
    for (int j = 0; j < 4; ++j) { v[j] = expf(v[j] - mx); sum += v[j]; }
    red[t] = sum; __syncthreads();
    for (int st = 128; st > 0; st >>= 1) { if (t < st) red[t] += red[t + st]; __syncthreads(); }
    const float inv = 1.f / red[0];
    #pragma unroll
    for (int j = 0; j < 4; ++j) {
        const float o = v[j] * inv;
        r[t + j * 256] = o;
        sb[base + t + j * 256] = (bf16_t)o;
    }
}

// V (NB, SRC, 512) f32 -> Vt (NB, 512, SRC) bf16
__global__ __launch_bounds__(256)
void k_transpose_v(const float* __restrict__ v, bf16_t* __restrict__ vt)
{
    const long i = (long)blockIdx.x * 256 + threadIdx.x;
    const int l = (int)(i & (SRC_C - 1));
    const long bc = i >> 10;
    const int c = (int)(bc & 511);
    const long b = bc >> 9;
    vt[i] = (bf16_t)v[((b * SRC_C + l) << 9) + c];
}

// ---------------------------------------------------------------------------
// Host orchestration
// ---------------------------------------------------------------------------
static inline void gemm(hipStream_t s, const bf16_t* A, const bf16_t* B,
                        const float* bias, float* C, bf16_t* Cbf,
                        int M, int N, int K, int lda, int ldb, int ldc,
                        int act, int accum, int batches = 1,
                        long sAb = 0, long sAh = 0, long sBb = 0, long sBh = 0,
                        long sCb = 0, long sCh = 0)
{
    const int tiles = (M / 32) * (N / 32);
    dim3 g((tiles + 7) / 8, batches);
    k_gemm<<<g, 256, 0, s>>>(A, B, bias, C, Cbf, M, N, K, lda, ldb, ldc, act, accum,
                             sAb, sAh, sBb, sBh, sCb, sCh);
}
static inline void cast_bf16(hipStream_t s, const float* in, bf16_t* out, long n)
{
    long blocks = (n + 255) / 256;
    if (blocks > 32768) blocks = 32768;
    k_cast<<<(int)blocks, 256, 0, s>>>(in, out, n);
}

struct Scratch {
    float *f_x, *f_y, *f_xz, *f_xc, *f_dbc, *f_dt, *f_ys;
    float *f_q, *f_k, *f_v, *f_sc, *f_ffn;
    bf16_t *h_a, *h_b, *h_w, *h_dbc, *h_mem, *h_q, *h_k, *h_vt, *h_s;
};

static void mamba_layer(hipStream_t s, const Scratch& w, void* const* d_in,
                        int base, float* x, int L)
{
    const int M = NB_C * L;
    auto F = [&](int i) { return (const float*)d_in[i]; };
    const float *g = F(base + 0), *b = F(base + 1);
    const float *in_w = F(base + 2), *conv_w = F(base + 3), *conv_b = F(base + 4);
    const float *xproj_w = F(base + 5), *dt_w = F(base + 6), *dt_b = F(base + 7);
    const float *A_log = F(base + 8), *Dp = F(base + 9), *out_w = F(base + 10);

    k_layernorm<<<M, 256, 0, s>>>(x, g, b, w.h_a);
    cast_bf16(s, in_w, w.h_w, 2048L * 512);
    gemm(s, w.h_a, w.h_w, nullptr, w.f_xz, nullptr, M, 2048, 512, 512, 512, 2048, 0, 0);

    const long nconv = (long)M * D_INNER_C;
    k_conv_silu<<<(int)((nconv + 255) / 256), 256, 0, s>>>(w.f_xz, conv_w, conv_b,
                                                           w.f_xc, w.h_a, L, nconv);

    cast_bf16(s, xproj_w, w.h_w, 64L * 1024);
    gemm(s, w.h_a, w.h_w, nullptr, w.f_dbc, w.h_dbc, M, 64, 1024, 1024, 1024, 64, 0, 0);

    cast_bf16(s, dt_w, w.h_w, 1024L * 32);
    gemm(s, w.h_dbc, w.h_w, dt_b, w.f_dt, nullptr, M, 1024, 32, 64, 32, 1024, 2, 0);

    k_scan<<<dim3(D_INNER_C / 256, NB_C), 256, 0, s>>>(w.f_dt, w.f_dbc, w.f_xc, A_log, Dp, w.f_ys, L);
    k_gate<<<(int)((nconv + 255) / 256), 256, 0, s>>>(w.f_ys, w.f_xz, w.h_a, nconv);

    cast_bf16(s, out_w, w.h_w, 512L * 1024);
    gemm(s, w.h_a, w.h_w, nullptr, x, nullptr, M, 512, 1024, 1024, 1024, 512, 0, 1); // residual
}

extern "C" void kernel_launch(void* const* d_in, const int* in_sizes, int n_in,
                              void* d_out, int out_size, void* d_ws, size_t ws_size,
                              hipStream_t stream)
{
    (void)in_sizes; (void)n_in; (void)out_size; (void)ws_size;
    auto F = [&](int i) { return (const float*)d_in[i]; };
    const int* src_ids = (const int*)d_in[0];
    const int* tgt_ids = (const int*)d_in[1];
    const float* emb = F(2);
    const float *enc_ng = F(231), *enc_nb = F(232), *dec_ng = F(233), *dec_nb = F(234);

    char* ws = (char*)d_ws;
    size_t off = 0;
    auto alloc = [&](size_t bytes) -> char* {
        char* p = ws + off;
        off += (bytes + 255) & ~(size_t)255;
        return p;
    };
    Scratch w;
    w.f_x   = (float*)alloc((size_t)MENC_C * 512 * 4);
    w.f_y   = (float*)alloc((size_t)MDEC_C * 512 * 4);
    w.f_xz  = (float*)alloc((size_t)MENC_C * 2048 * 4);
    w.f_xc  = (float*)alloc((size_t)MENC_C * 1024 * 4);
    w.f_dbc = (float*)alloc((size_t)MENC_C * 64 * 4);
    w.f_dt  = (float*)alloc((size_t)MENC_C * 1024 * 4);
    w.f_ys  = (float*)alloc((size_t)MENC_C * 1024 * 4);
    w.f_q   = (float*)alloc((size_t)MDEC_C * 512 * 4);
    w.f_k   = (float*)alloc((size_t)MENC_C * 512 * 4);
    w.f_v   = (float*)alloc((size_t)MENC_C * 512 * 4);
    w.f_sc  = (float*)alloc((size_t)16 * TGT_C * SRC_C * 4);
    w.f_ffn = (float*)alloc((size_t)MDEC_C * 2048 * 4);
    w.h_a   = (bf16_t*)alloc((size_t)MENC_C * 2048 * 2);
    w.h_b   = (bf16_t*)alloc((size_t)MENC_C * 2048 * 2);
    w.h_w   = (bf16_t*)alloc((size_t)VOCAB_C * 512 * 2);
    w.h_dbc = (bf16_t*)alloc((size_t)MENC_C * 64 * 2);
    w.h_mem = (bf16_t*)alloc((size_t)MENC_C * 512 * 2);
    w.h_q   = (bf16_t*)alloc((size_t)MDEC_C * 512 * 2);
    w.h_k   = (bf16_t*)alloc((size_t)MENC_C * 512 * 2);
    w.h_vt  = (bf16_t*)alloc((size_t)NB_C * 512 * SRC_C * 2);
    w.h_s   = (bf16_t*)alloc((size_t)16 * TGT_C * SRC_C * 2);

    // ---------------- Encoder ----------------
    k_embed<<<MENC_C, 256, 0, stream>>>(src_ids, emb, w.f_x);
    for (int i = 0; i < 6; ++i)
        mamba_layer(stream, w, d_in, 3 + 11 * i, w.f_x, SRC_C);
    k_layernorm<<<MENC_C, 256, 0, stream>>>(w.f_x, enc_ng, enc_nb, w.h_mem);

    // ---------------- Decoder ----------------
    k_embed<<<MDEC_C, 256, 0, stream>>>(tgt_ids, emb, w.f_y);
    for (int i = 0; i < 6; ++i) {
        const int base = 69 + 27 * i;
        mamba_layer(stream, w, d_in, base, w.f_y, TGT_C);

        // cross-attention
        const float *cn_g = F(base + 11), *cn_b = F(base + 12);
        const float *q_w = F(base + 13), *q_b = F(base + 14);
        const float *k_w = F(base + 15), *k_b = F(base + 16);
        const float *v_w = F(base + 17), *v_b = F(base + 18);
        const float *o_w = F(base + 19), *o_b = F(base + 20);

        k_layernorm<<<MDEC_C, 256, 0, stream>>>(w.f_y, cn_g, cn_b, w.h_a);
        cast_bf16(stream, q_w, w.h_w, 512L * 512);
        gemm(stream, w.h_a, w.h_w, q_b, w.f_q, w.h_q, MDEC_C, 512, 512, 512, 512, 512, 0, 0);
        cast_bf16(stream, k_w, w.h_w, 512L * 512);
        gemm(stream, w.h_mem, w.h_w, k_b, w.f_k, w.h_k, MENC_C, 512, 512, 512, 512, 512, 0, 0);
        cast_bf16(stream, v_w, w.h_w, 512L * 512);
        gemm(stream, w.h_mem, w.h_w, v_b, w.f_v, nullptr, MENC_C, 512, 512, 512, 512, 512, 0, 0);
        k_transpose_v<<<(NB_C * 512 * SRC_C) / 256, 256, 0, stream>>>(w.f_v, w.h_vt);

        // scores[b,h] = Q[b,:,h,:] @ K[b,:,h,:]^T   (batched over gridDim.y=16)
        gemm(stream, w.h_q, w.h_k, nullptr, w.f_sc, nullptr,
             TGT_C, SRC_C, HD_C, 512, 512, SRC_C, 0, 0, 16,
             (long)TGT_C * 512, HD_C,                 // A: b, h strides
             (long)SRC_C * 512, HD_C,                 // B: b, h strides
             8L * TGT_C * SRC_C, (long)TGT_C * SRC_C);// C: b, h strides

        k_softmax<<<16 * TGT_C, 256, 0, stream>>>(w.f_sc, w.h_s);

        // attout[b,:,h,:] = softmax(scores) @ V[b,:,h,:]
        gemm(stream, w.h_s, w.h_vt, nullptr, w.f_q, w.h_a,
             TGT_C, HD_C, SRC_C, SRC_C, SRC_C, 512, 0, 0, 16,
             8L * TGT_C * SRC_C, (long)TGT_C * SRC_C, // A: b, h strides
             512L * SRC_C, (long)HD_C * SRC_C,        // B: b, h strides
             (long)TGT_C * 512, HD_C);                // C: b, h strides

        cast_bf16(stream, o_w, w.h_w, 512L * 512);
        gemm(stream, w.h_a, w.h_w, o_b, w.f_y, nullptr, MDEC_C, 512, 512, 512, 512, 512, 0, 1);

        // FFN
        const float *fn_g = F(base + 21), *fn_b = F(base + 22);
        const float *ff1_w = F(base + 23), *ff1_b = F(base + 24);
        const float *ff2_w = F(base + 25), *ff2_b = F(base + 26);
        k_layernorm<<<MDEC_C, 256, 0, stream>>>(w.f_y, fn_g, fn_b, w.h_a);
        cast_bf16(stream, ff1_w, w.h_w, 2048L * 512);
        gemm(stream, w.h_a, w.h_w, ff1_b, w.f_ffn, w.h_b, MDEC_C, 2048, 512, 512, 512, 2048, 3, 0);
        cast_bf16(stream, ff2_w, w.h_w, 512L * 2048);
        gemm(stream, w.h_b, w.h_w, ff2_b, w.f_y, nullptr, MDEC_C, 512, 2048, 2048, 2048, 512, 0, 1);
    }

    // ---------------- Final projection to vocab ----------------
    k_layernorm<<<MDEC_C, 256, 0, stream>>>(w.f_y, dec_ng, dec_nb, w.h_a);
    cast_bf16(stream, emb, w.h_w, (long)VOCAB_C * 512);
    gemm(stream, w.h_a, w.h_w, nullptr, (float*)d_out, nullptr,
         MDEC_C, VOCAB_C, 512, 512, 512, VOCAB_C, 0, 0);
}
